// Model2_16904991277618
// MI455X (gfx1250) — compile-verified, matrix operand order checked by reading, not run on hardware
//
#include <hip/hip_runtime.h>
#include <math.h>

typedef __attribute__((ext_vector_type(16))) __bf16 v16bf;
typedef __attribute__((ext_vector_type(8)))  float  v8f;
typedef __attribute__((ext_vector_type(4)))  float  v4f;   // native vector for nontemporal b128 loads

#define Bsz 512
#define Tn  1024
#define Fn  64
#define Hn  40
#define G4  160    // 4*H
#define KPAD 64
#define NTHREADS 320   // 10 wave32 -> 10 N-tiles of 16 gate columns

// Branch-free activations: v_exp_f32 + raw v_rcp_f32 (no IEEE div expansion).
__device__ __forceinline__ float rcp_(float x){ return __builtin_amdgcn_rcpf(x); }
__device__ __forceinline__ float sigmoidf_(float x){ return rcp_(1.0f + __expf(-x)); }
__device__ __forceinline__ float tanhf_(float x){ return 2.0f * rcp_(1.0f + __expf(-2.0f * x)) - 1.0f; }

__device__ __forceinline__ v8f wmma_bf16(v16bf a, v16bf b, v8f c){
  // D(f32,16x16) = A(bf16,16x32) * B(bf16,32x16) + C
  return __builtin_amdgcn_wmma_f32_16x16x32_bf16(false, a, false, b, (short)0, c, false, false);
}

// A-fragment (16x32 bf16) from an LDS tile stored row-major [16][KPAD].
// ISA layout: lane m=lane&15, kh=lane>>4; elems 0..7 -> K=kh*8+0..7, elems 8..15 -> K=16+kh*8+0..7
__device__ __forceinline__ v16bf ldsA(const __bf16* S, int k0){
  int lane = threadIdx.x & 31;
  int m  = lane & 15;
  int kh = (lane >> 4) & 1;
  const __bf16* p = S + m * KPAD + k0 + kh * 8;
  v16bf a;
  #pragma unroll
  for (int e = 0; e < 8; ++e) a[e] = p[e];
  #pragma unroll
  for (int e = 0; e < 8; ++e) a[8 + e] = p[16 + e];
  return a;
}

// B-fragment (32x16 bf16) from global f32 weight W[K][N] row-major, zero-padding k >= K.
__device__ __forceinline__ v16bf gB(const float* __restrict__ W, int K, int N, int k0, int n0){
  int lane = threadIdx.x & 31;
  int n  = lane & 15;
  int kh = (lane >> 4) & 1;
  v16bf b;
  #pragma unroll
  for (int e = 0; e < 16; ++e){
    int k  = (e < 8) ? (kh * 8 + e) : (16 + kh * 8 + (e - 8));
    int kk = k0 + k;
    float v = (kk < K) ? W[kk * N + n0 + n] : 0.0f;
    b[e] = (__bf16)v;
  }
  return b;
}

__global__ __launch_bounds__(NTHREADS)
void lstm2_fused(const float* __restrict__ seq,  const float* __restrict__ feat,
                 const float* __restrict__ Wa_k, const float* __restrict__ Wa_r, const float* __restrict__ ba,
                 const float* __restrict__ Wb_k, const float* __restrict__ Wb_r, const float* __restrict__ bb,
                 const float* __restrict__ Wg,   const float* __restrict__ bg,
                 const float* __restrict__ Wh,   const float* __restrict__ bh,
                 const float* __restrict__ Wc,   const float* __restrict__ bc,
                 const float* __restrict__ Wd,   const float* __restrict__ bd,
                 const float* __restrict__ Wo,   const float* __restrict__ bo,
                 float* __restrict__ out)
{
  __shared__ alignas(16) __bf16 xbuf[2][16 * KPAD];   // double-buffered x_t tile (bf16)
  __shared__ alignas(16) __bf16 hA[16 * KPAD];        // LSTM-A hidden, zero-padded K
  __shared__ alignas(16) __bf16 hB[16 * KPAD];        // LSTM-B hidden, zero-padded K
  __shared__ float zbuf[16 * G4];                     // pre-activation gates [16][160]
  __shared__ float hBf[16 * Hn];                      // final h_b in f32 for the tail
  __shared__ float y1[16 * 10], y2[16 * 10], cbv[16 * 20], dbv[16 * 10];

  const int tid  = threadIdx.x;
  const int wave = tid >> 5;        // 0..9
  const int n0   = wave * 16;       // gate-column tile
  const int lane = tid & 31;
  const int b0   = blockIdx.x * 16; // batch-row tile

  // zero-init hidden state (including K padding 40..63)
  for (int i = tid; i < 16 * KPAD; i += NTHREADS){ hA[i] = (__bf16)0.0f; hB[i] = (__bf16)0.0f; }

  // Resident weight fragments (constant over the whole scan): 64 VGPRs/wave
  v16bf Bak0 = gB(Wa_k, Fn, G4, 0,  n0), Bak1 = gB(Wa_k, Fn, G4, 32, n0);
  v16bf Bar0 = gB(Wa_r, Hn, G4, 0,  n0), Bar1 = gB(Wa_r, Hn, G4, 32, n0);
  v16bf Bbk0 = gB(Wb_k, Hn, G4, 0,  n0), Bbk1 = gB(Wb_k, Hn, G4, 32, n0);
  v16bf Bbr0 = gB(Wb_r, Hn, G4, 0,  n0), Bbr1 = gB(Wb_r, Hn, G4, 32, n0);
  const float baL = ba[n0 + (lane & 15)];
  const float bbL = bb[n0 + (lane & 15)];

  // Cell state: each thread owns exactly two (row, j) gate elements (640 = 2*320)
  float cA[2] = {0.f, 0.f}, cB[2] = {0.f, 0.f};
  int rowi[2], ji[2];
  #pragma unroll
  for (int q = 0; q < 2; ++q){ int idx = tid + q * NTHREADS; rowi[q] = idx / Hn; ji[q] = idx % Hn; }

  // x prefetch assignment: threads 0..255 each own one v4f (256*4 = 16*64)
  const int xrow = tid >> 4;            // 0..15 (valid when tid < 256)
  const int xcol = (tid & 15) * 4;      // 0,4,...,60

  // stage x_0
  if (tid < 256){
    const v4f v = __builtin_nontemporal_load(
        (const v4f*)&seq[((size_t)(b0 + xrow) * Tn + 0) * Fn + xcol]);
    __bf16* p = &xbuf[0][xrow * KPAD + xcol];
    p[0] = (__bf16)v.x; p[1] = (__bf16)v.y; p[2] = (__bf16)v.z; p[3] = (__bf16)v.w;
  }
  __syncthreads();

  v4f xr;
  for (int t = 0; t < Tn; ++t){
    const int buf = t & 1;

    // prefetch x_{t+1} into registers (hides HBM latency behind this step's compute)
    if (t + 1 < Tn && tid < 256){
      xr = __builtin_nontemporal_load(
          (const v4f*)&seq[((size_t)(b0 + xrow) * Tn + (t + 1)) * Fn + xcol]);
    }

    // ---- LSTM A: z = x_t @ Wa_k + h_a @ Wa_r + ba ----
    {
      v8f acc;
      #pragma unroll
      for (int r = 0; r < 8; ++r) acc[r] = baL;
      acc = wmma_bf16(ldsA(xbuf[buf], 0),  Bak0, acc);
      acc = wmma_bf16(ldsA(xbuf[buf], 32), Bak1, acc);
      acc = wmma_bf16(ldsA(hA, 0),  Bar0, acc);
      acc = wmma_bf16(ldsA(hA, 32), Bar1, acc);
      int mh = (lane >> 4) & 1, nn = lane & 15;
      #pragma unroll
      for (int r = 0; r < 8; ++r) zbuf[(r + 8 * mh) * G4 + n0 + nn] = acc[r];
    }
    __syncthreads();

    // ---- gates A (i,f,g,o), update c_a / h_a ----
    #pragma unroll
    for (int q = 0; q < 2; ++q){
      const float* zr = &zbuf[rowi[q] * G4];
      float ig = sigmoidf_(zr[ji[q]]);
      float fg = sigmoidf_(zr[Hn + ji[q]]);
      float gg = tanhf_(zr[2 * Hn + ji[q]]);
      float og = sigmoidf_(zr[3 * Hn + ji[q]]);
      float c  = fg * cA[q] + ig * gg;
      cA[q] = c;
      hA[rowi[q] * KPAD + ji[q]] = (__bf16)(og * tanhf_(c));
    }
    // stage prefetched x into the other LDS buffer
    if (t + 1 < Tn && tid < 256){
      __bf16* p = &xbuf[1 - buf][xrow * KPAD + xcol];
      p[0] = (__bf16)xr.x; p[1] = (__bf16)xr.y; p[2] = (__bf16)xr.z; p[3] = (__bf16)xr.w;
    }
    __syncthreads();

    // ---- LSTM B: z = h_a @ Wb_k + h_b @ Wb_r + bb ----
    {
      v8f acc;
      #pragma unroll
      for (int r = 0; r < 8; ++r) acc[r] = bbL;
      acc = wmma_bf16(ldsA(hA, 0),  Bbk0, acc);
      acc = wmma_bf16(ldsA(hA, 32), Bbk1, acc);
      acc = wmma_bf16(ldsA(hB, 0),  Bbr0, acc);
      acc = wmma_bf16(ldsA(hB, 32), Bbr1, acc);
      int mh = (lane >> 4) & 1, nn = lane & 15;
      #pragma unroll
      for (int r = 0; r < 8; ++r) zbuf[(r + 8 * mh) * G4 + n0 + nn] = acc[r];
    }
    __syncthreads();

    // ---- gates B, update c_b / h_b ----
    #pragma unroll
    for (int q = 0; q < 2; ++q){
      const float* zr = &zbuf[rowi[q] * G4];
      float ig = sigmoidf_(zr[ji[q]]);
      float fg = sigmoidf_(zr[Hn + ji[q]]);
      float gg = tanhf_(zr[2 * Hn + ji[q]]);
      float og = sigmoidf_(zr[3 * Hn + ji[q]]);
      float c  = fg * cB[q] + ig * gg;
      cB[q] = c;
      float h = og * tanhf_(c);
      hB[rowi[q] * KPAD + ji[q]] = (__bf16)h;
      if (t == Tn - 1) hBf[rowi[q] * Hn + ji[q]] = h;
    }
    __syncthreads();
  }

  // ---- dense tail (tiny, scalar f32) ----
  if (tid < 160){                         // y1 = tanh(feat @ Wg + bg)
    int r = tid / 10, d = tid % 10;
    float s = bg[d];
    for (int f = 0; f < Fn; ++f) s += feat[(size_t)(b0 + r) * Fn + f] * Wg[f * 10 + d];
    y1[r * 10 + d] = tanhf_(s);
  }
  __syncthreads();
  if (tid < 160){                         // y2 = tanh(y1 @ Wh + bh)
    int r = tid / 10, d = tid % 10;
    float s = bh[d];
    for (int k = 0; k < 10; ++k) s += y1[r * 10 + k] * Wh[k * 10 + d];
    y2[r * 10 + d] = tanhf_(s);
  }
  __syncthreads();
  {                                       // c = relu([h_b | y2] @ Wc + bc)  (16*20 == 320 threads)
    int r = tid / 20, k = tid % 20;
    float s = bc[k];
    for (int j = 0; j < Hn; ++j) s += hBf[r * Hn + j] * Wc[j * 20 + k];
    for (int j = 0; j < 10; ++j) s += y2[r * 10 + j] * Wc[(Hn + j) * 20 + k];
    cbv[r * 20 + k] = fmaxf(s, 0.0f);
  }
  __syncthreads();
  if (tid < 160){                         // d = relu(c @ Wd + bd)
    int r = tid / 10, k = tid % 10;
    float s = bd[k];
    for (int j = 0; j < 20; ++j) s += cbv[r * 20 + j] * Wd[j * 10 + k];
    dbv[r * 10 + k] = fmaxf(s, 0.0f);
  }
  __syncthreads();
  if (tid < 16){                          // out = sigmoid(d @ Wo + bo)
    float s = bo[0];
    for (int j = 0; j < 10; ++j) s += dbv[tid * 10 + j] * Wo[j];
    out[b0 + tid] = sigmoidf_(s);
  }
}

extern "C" void kernel_launch(void* const* d_in, const int* in_sizes, int n_in,
                              void* d_out, int out_size, void* d_ws, size_t ws_size,
                              hipStream_t stream) {
  const float* seq  = (const float*)d_in[0];
  const float* feat = (const float*)d_in[1];
  const float* Wa_k = (const float*)d_in[2];
  const float* Wa_r = (const float*)d_in[3];
  const float* ba   = (const float*)d_in[4];
  const float* Wb_k = (const float*)d_in[5];
  const float* Wb_r = (const float*)d_in[6];
  const float* bb   = (const float*)d_in[7];
  const float* Wg   = (const float*)d_in[8];
  const float* bg   = (const float*)d_in[9];
  const float* Wh   = (const float*)d_in[10];
  const float* bh   = (const float*)d_in[11];
  const float* Wc   = (const float*)d_in[12];
  const float* bc   = (const float*)d_in[13];
  const float* Wd   = (const float*)d_in[14];
  const float* bd   = (const float*)d_in[15];
  const float* Wo   = (const float*)d_in[16];
  const float* bo   = (const float*)d_in[17];
  float* out = (float*)d_out;

  lstm2_fused<<<Bsz / 16, NTHREADS, 0, stream>>>(
      seq, feat, Wa_k, Wa_r, ba, Wb_k, Wb_r, bb,
      Wg, bg, Wh, bh, Wc, bc, Wd, bd, Wo, bo, out);
}
